// Quantizer_84799834293036
// MI455X (gfx1250) — matrix operand order repaired and checked by reading, not compiled, not run on hardware
//
#include <hip/hip_runtime.h>

typedef __bf16 v16bf __attribute__((ext_vector_type(16)));
typedef float v8f __attribute__((ext_vector_type(8)));

#define NTOK 16384
#define DIM 256
#define NE 8192

// ---- d_out layout (float elements) ----
#define OFF_QUANT  0
#define OFF_DIFF   (NTOK * DIM)                 // 4194304
#define OFF_IND    (OFF_DIFF + 1)               // 4194305
#define OFF_ONEHOT (OFF_IND + NTOK)             // 4210689
#define OFF_ESUM   (OFF_ONEHOT + NE)            // 4218881
#define N_ZERO     (NE + NE * DIM)              // onehot + esum contiguous

// ---- workspace layout (byte offsets) ----
#define WS_A     ((size_t)0)                    // bf16 [NTOK][DIM]   8 MB
#define WS_BT    ((size_t)8u  << 20)            // bf16 [NE][DIM]     4 MB
#define WS_ET    ((size_t)12u << 20)            // f32  [NE][DIM]     8 MB
#define WS_ENORM ((size_t)20u << 20)            // f32  [NE]
#define WS_IND   (WS_ENORM + 32768)             // int  [NTOK]
#define WS_DIFF  (WS_IND + 65536)               // f32  [1]

__device__ __forceinline__ unsigned int f2bf(float f) {
    unsigned int u = __float_as_uint(f);
    unsigned int r = u + 0x7FFFu + ((u >> 16) & 1u);   // round-to-nearest-even
    return r >> 16;
}

// ------------------------------------------------------------------
// Kernel 0: zero the accumulated outputs + diff accumulator
// ------------------------------------------------------------------
__global__ void k_zero(float* __restrict__ out, float* __restrict__ wsDiff) {
    int i = blockIdx.x * blockDim.x + threadIdx.x;
    if (i < N_ZERO) out[OFF_ONEHOT + i] = 0.0f;
    if (i == 0) wsDiff[0] = 0.0f;
}

// ------------------------------------------------------------------
// Kernel 1: x (f32, row-major [NTOK][DIM]) -> A (bf16, same layout)
// One wave per token row.
// ------------------------------------------------------------------
__global__ __launch_bounds__(256) void k_prep_a(const float* __restrict__ x,
                                                unsigned short* __restrict__ A) {
    const int wid = threadIdx.x >> 5, lane = threadIdx.x & 31;
    const int row = blockIdx.x * 8 + wid;
    const float4* xp = (const float4*)(x + (size_t)row * DIM) + lane * 2;
    float4 a = xp[0], b = xp[1];
    uint4 w;
    w.x = f2bf(a.x) | (f2bf(a.y) << 16);
    w.y = f2bf(a.z) | (f2bf(a.w) << 16);
    w.z = f2bf(b.x) | (f2bf(b.y) << 16);
    w.w = f2bf(b.z) | (f2bf(b.w) << 16);
    ((uint4*)A)[(size_t)row * 32 + lane] = w;
}

// ------------------------------------------------------------------
// Kernel 2: embed [DIM][NE] f32 -> Bt [NE][DIM] bf16 and eT [NE][DIM] f32
// LDS-tiled 32x32 transpose.
// ------------------------------------------------------------------
__global__ __launch_bounds__(256) void k_transpose(const float* __restrict__ embed,
                                                   unsigned short* __restrict__ Bt,
                                                   float* __restrict__ eT) {
    __shared__ float tile[32][33];
    const int k0 = blockIdx.x * 32;   // gridDim.x = DIM/32 = 8
    const int j0 = blockIdx.y * 32;   // gridDim.y = NE/32  = 256
    const int tx = threadIdx.x & 31, tg = threadIdx.x >> 5;
#pragma unroll
    for (int i = 0; i < 4; ++i) {
        int k = tg + i * 8;
        tile[k][tx] = embed[(size_t)(k0 + k) * NE + j0 + tx];
    }
    __syncthreads();
    const int j = threadIdx.x >> 3;   // 0..31
    const int q = threadIdx.x & 7;    // 0..7 -> 4 k's each
    float v0 = tile[q * 4 + 0][j], v1 = tile[q * 4 + 1][j];
    float v2 = tile[q * 4 + 2][j], v3 = tile[q * 4 + 3][j];
    float4 fv = make_float4(v0, v1, v2, v3);
    *(float4*)(eT + (size_t)(j0 + j) * DIM + k0 + q * 4) = fv;
    uint2 bv;
    bv.x = f2bf(v0) | (f2bf(v1) << 16);
    bv.y = f2bf(v2) | (f2bf(v3) << 16);
    *(uint2*)(Bt + (size_t)(j0 + j) * DIM + k0 + q * 4) = bv;
}

// ------------------------------------------------------------------
// Kernel 3: enorm[j] = sum_k embed[k][j]^2   (coalesced over j)
// ------------------------------------------------------------------
__global__ __launch_bounds__(256) void k_enorm(const float* __restrict__ embed,
                                               float* __restrict__ enorm) {
    const int j = blockIdx.x * 256 + threadIdx.x;
    float s = 0.0f;
#pragma unroll 8
    for (int k = 0; k < DIM; ++k) {
        float e = embed[(size_t)k * NE + j];
        s += e * e;
    }
    enorm[j] = s;
}

// ------------------------------------------------------------------
// Kernel 4: WMMA GEMM (bf16 -> f32 acc) fused with per-row argmin.
// 256 WGs x 128 threads (4 waves); each wave owns 16 rows, loops all
// 8192 codes in 64-column LDS chunks.
// B chunks staged with GLOBAL_LOAD_ASYNC_TO_LDS_B128 (no VGPR staging,
// ASYNCcnt-tracked), double-buffered in LDS.
// ------------------------------------------------------------------
#define GEMM_BLOCK 128
#define ROWS_PER_WG 64
#define CHUNK_COLS 64
#define NCHUNK (NE / CHUNK_COLS)   // 128
#define BSTRIDE 264                // padded (bf16 elems) -> conflict-light b128
#define CHUNK_BYTES (CHUNK_COLS * DIM * 2)   // 32768 B of Bt per chunk

union Frag { uint4 u[2]; v16bf v; };

// Each thread issues 16 async b128s; INST_OFFSET advances BOTH the global and
// the LDS address (ISA 10.7/15.18.3), so one base pair covers 256 bytes.
__device__ __forceinline__ void issue_chunk_async(unsigned ldsaddr, const char* g) {
#pragma unroll
    for (int p = 0; p < 16; ++p)
        asm volatile("global_load_async_to_lds_b128 %0, %1, off offset:%2"
                     :: "v"(ldsaddr), "v"(g), "i"(p * 16)
                     : "memory");
}

__global__ __launch_bounds__(GEMM_BLOCK) void k_gemm_argmin(
    const unsigned short* __restrict__ A, const unsigned short* __restrict__ Bt,
    const float* __restrict__ enorm, int* __restrict__ wsInd,
    float* __restrict__ outIndF, float* __restrict__ outOnehot) {
    __shared__ __align__(16) unsigned short Bs[2][CHUNK_COLS * BSTRIDE];  // 66 KB

    const int tid = threadIdx.x;
    const int wid = tid >> 5;
    const int lane = tid & 31;
    const int m = lane & 15;     // row within tile (A) / col within tile (B,C)
    const int hh = lane >> 4;    // half selector per ISA fragment layout

    const int rblk = blockIdx.x * ROWS_PER_WG + wid * 16;

    // A fragments: 16 rows x K=256 resident in registers (8 frags x 8 VGPRs).
    // 16-bit A layout: lo u4 = K[hh*8 .. +7], hi u4 = K[16+hh*8 .. +7].
    Frag af[8];
    {
        const uint4* Ap = (const uint4*)A + (size_t)(rblk + m) * 32 + hh;
#pragma unroll
        for (int ks = 0; ks < 8; ++ks) {
            af[ks].u[0] = Ap[ks * 4];
            af[ks].u[1] = Ap[ks * 4 + 2];
        }
    }

    float best[8];
    int bidx[8];
#pragma unroll
    for (int v = 0; v < 8; ++v) { best[v] = 3.4e38f; bidx[v] = 0; }

    // async staging: 128 threads x 256B = 64 cols x 512B per chunk
    const int scol = tid >> 1;
    const int sh = tid & 1;
    // LDS byte address = low 32 bits of the generic address (LDS aperture).
    const unsigned lds0 = (unsigned)(uintptr_t)&Bs[0][scol * BSTRIDE + sh * 128];
    const unsigned lds1 = (unsigned)(uintptr_t)&Bs[1][scol * BSTRIDE + sh * 128];
    const char* gbase = (const char*)Bt + ((size_t)scol * DIM + sh * 128) * 2;

    issue_chunk_async(lds0, gbase);  // chunk 0 -> buf 0

    for (int chunk = 0; chunk < NCHUNK; ++chunk) {
        const int b = chunk & 1;
        if (chunk + 1 < NCHUNK) {
            issue_chunk_async(b ? lds0 : lds1,
                              gbase + (size_t)(chunk + 1) * CHUNK_BYTES);
            // 16 newer ops outstanding; my 16 chunk-`chunk` ops are older and
            // ASYNC completions are in-order -> wait <=16 makes them visible.
            asm volatile("s_wait_asynccnt 0x10" ::: "memory");
        } else {
            asm volatile("s_wait_asynccnt 0x0" ::: "memory");
        }
        __syncthreads();   // all waves' staged data in LDS
#pragma unroll
        for (int ct = 0; ct < 4; ++ct) {
            v8f acc = {0.f, 0.f, 0.f, 0.f, 0.f, 0.f, 0.f, 0.f};
            const unsigned short* bp = &Bs[b][(ct * 16 + m) * BSTRIDE + hh * 8];
#pragma unroll
            for (int ks = 0; ks < 8; ++ks) {
                Frag bf_;
                bf_.u[0] = *(const uint4*)(bp + ks * 32);
                bf_.u[1] = *(const uint4*)(bp + ks * 32 + 16);
                acc = __builtin_amdgcn_wmma_f32_16x16x32_bf16(
                    false, af[ks].v, false, bf_.v, (short)0, acc, false, false);
            }
            const int colg = chunk * CHUNK_COLS + ct * 16 + m;
            const float en = enorm[colg];
#pragma unroll
            for (int v = 0; v < 8; ++v) {
                const float s = en - 2.0f * acc[v];    // argmin-equivalent score
                if (s < best[v]) { best[v] = s; bidx[v] = colg; }
            }
        }
        __syncthreads();   // all waves done reading buf b before it is refilled
    }

    // Butterfly argmin across the 16 lanes holding each row's columns.
    // Offsets <16 keep lanes 0-15 (rows v) and 16-31 (rows v+8) independent.
#pragma unroll
    for (int v = 0; v < 8; ++v) {
        float s = best[v];
        int i = bidx[v];
        for (int off = 8; off >= 1; off >>= 1) {
            float so = __shfl_xor(s, off, 32);
            int io = __shfl_xor(i, off, 32);
            if (so < s || (so == s && io < i)) { s = so; i = io; }  // first-index ties
        }
        best[v] = s;
        bidx[v] = i;
    }
    if (m == 0) {  // lanes 0 and 16
#pragma unroll
        for (int v = 0; v < 8; ++v) {
            const int row = rblk + hh * 8 + v;
            wsInd[row] = bidx[v];
            outIndF[row] = (float)bidx[v];
            atomicAdd(&outOnehot[bidx[v]], 1.0f);
        }
    }
}

// ------------------------------------------------------------------
// Kernel 5: gather quantize (== quantize_st forward value), diff partials,
// embed_sum scatter. One wave per token.
// ------------------------------------------------------------------
__global__ __launch_bounds__(256) void k_gather(const float* __restrict__ x,
                                                const float* __restrict__ eT,
                                                const int* __restrict__ wsInd,
                                                float* __restrict__ outQ,
                                                float* __restrict__ outEsum,
                                                float* __restrict__ wsDiff) {
    const int wid = threadIdx.x >> 5, lane = threadIdx.x & 31;
    const int tok = blockIdx.x * 8 + wid;
    const int code = wsInd[tok];
    const float4* ep = (const float4*)(eT + (size_t)code * DIM) + lane * 2;
    const float4* xp = (const float4*)(x + (size_t)tok * DIM) + lane * 2;
    float4 q0 = ep[0], q1 = ep[1];
    float4 x0 = xp[0], x1 = xp[1];
    float4* op = (float4*)(outQ + (size_t)tok * DIM) + lane * 2;
    op[0] = q0;
    op[1] = q1;
    float d = (q0.x - x0.x) * (q0.x - x0.x) + (q0.y - x0.y) * (q0.y - x0.y) +
              (q0.z - x0.z) * (q0.z - x0.z) + (q0.w - x0.w) * (q0.w - x0.w) +
              (q1.x - x1.x) * (q1.x - x1.x) + (q1.y - x1.y) * (q1.y - x1.y) +
              (q1.z - x1.z) * (q1.z - x1.z) + (q1.w - x1.w) * (q1.w - x1.w);
    for (int off = 16; off >= 1; off >>= 1) d += __shfl_xor(d, off, 32);
    if (lane == 0) atomicAdd(wsDiff, d);
    float xv[8] = {x0.x, x0.y, x0.z, x0.w, x1.x, x1.y, x1.z, x1.w};
#pragma unroll
    for (int u = 0; u < 8; ++u)
        atomicAdd(&outEsum[(size_t)(lane * 8 + u) * NE + code], xv[u]);
}

// ------------------------------------------------------------------
// Kernel 6: finalize diff mean
// ------------------------------------------------------------------
__global__ void k_finalize(const float* __restrict__ wsDiff, float* __restrict__ out) {
    out[OFF_DIFF] = wsDiff[0] * (1.0f / (float)(NTOK * DIM));
}

extern "C" void kernel_launch(void* const* d_in, const int* in_sizes, int n_in,
                              void* d_out, int out_size, void* d_ws, size_t ws_size,
                              hipStream_t stream) {
    const float* x = (const float*)d_in[0];       // [16,32,32,256] f32
    const float* embed = (const float*)d_in[1];   // [256,8192] f32
    float* out = (float*)d_out;

    char* ws = (char*)d_ws;
    unsigned short* A = (unsigned short*)(ws + WS_A);
    unsigned short* Bt = (unsigned short*)(ws + WS_BT);
    float* eT = (float*)(ws + WS_ET);
    float* enorm = (float*)(ws + WS_ENORM);
    int* wsInd = (int*)(ws + WS_IND);
    float* wsDiff = (float*)(ws + WS_DIFF);

    k_zero<<<(N_ZERO + 255) / 256, 256, 0, stream>>>(out, wsDiff);
    k_prep_a<<<NTOK / 8, 256, 0, stream>>>(x, A);
    k_transpose<<<dim3(DIM / 32, NE / 32), 256, 0, stream>>>(embed, Bt, eT);
    k_enorm<<<NE / 256, 256, 0, stream>>>(embed, enorm);
    k_gemm_argmin<<<NTOK / ROWS_PER_WG, GEMM_BLOCK, 0, stream>>>(
        A, Bt, enorm, wsInd, out + OFF_IND, out + OFF_ONEHOT);
    k_gather<<<NTOK / 8, 256, 0, stream>>>(x, eT, wsInd, out + OFF_QUANT,
                                           out + OFF_ESUM, wsDiff);
    k_finalize<<<1, 1, 0, stream>>>(wsDiff, out);
}
